// Uni3DPointcloudEncoder_9002251453216
// MI455X (gfx1250) — compile-verified, matrix operand order checked by reading, not run on hardware
//
#include <hip/hip_runtime.h>
#include <hip/hip_bf16.h>

#define B_   8
#define N_   8192
#define G_   512
#define K_   32
#define ENC_ 512
#define BG_  (B_ * G_)

typedef __attribute__((ext_vector_type(16))) _Float16 v16h;
typedef __attribute__((ext_vector_type(8)))  _Float16 v8h;
typedef __attribute__((ext_vector_type(8)))  float    v8f;

union V16 { v16h v; v8h h[2]; };

// ---------------------------------------------------------------------------
// 64-bit shuffle helper (wave32), built from 32-bit __shfl_xor
// ---------------------------------------------------------------------------
__device__ __forceinline__ unsigned long long shfl_xor_u64(unsigned long long v, int mask) {
    unsigned lo = (unsigned)v;
    unsigned hi = (unsigned)(v >> 32);
    lo = (unsigned)__shfl_xor((int)lo, mask, 32);
    hi = (unsigned)__shfl_xor((int)hi, mask, 32);
    return ((unsigned long long)hi << 32) | lo;
}

// ---------------------------------------------------------------------------
// Kernel: convert f32 weight [rows, scols] to f16 [rows, dcols], zero-padded
// ---------------------------------------------------------------------------
__global__ void cvt_pad_kernel(const float* __restrict__ src, _Float16* __restrict__ dst,
                               int rows, int scols, int dcols) {
    int i = blockIdx.x * 256 + threadIdx.x;
    if (i >= rows * dcols) return;
    int r = i / dcols, c = i - r * dcols;
    dst[i] = (c < scols) ? (_Float16)src[r * scols + c] : (_Float16)0.0f;
}

// ---------------------------------------------------------------------------
// Kernel: farthest point sampling. One block (1024 threads) per batch.
// min_d kept in registers (8 pts/thread); argmax via packed (dist|idx) keys.
// ---------------------------------------------------------------------------
__global__ void fps_kernel(const float* __restrict__ xyz, int* __restrict__ fidx) {
    const int b   = blockIdx.x;
    const int tid = threadIdx.x;          // 1024 threads
    const float* p = xyz + (size_t)b * N_ * 3;

    float px[8], py[8], pz[8], md[8];
#pragma unroll
    for (int i = 0; i < 8; ++i) {
        int pt = i * 1024 + tid;
        px[i] = p[pt * 3 + 0];
        py[i] = p[pt * 3 + 1];
        pz[i] = p[pt * 3 + 2];
        md[i] = 3.0e38f;
    }

    __shared__ float lx, ly, lz;
    __shared__ unsigned long long red[32];

    if (tid == 0) {
        lx = p[0]; ly = p[1]; lz = p[2];
        fidx[b * G_ + 0] = 0;
    }
    __syncthreads();

    for (int s = 1; s < G_; ++s) {
        float cx = lx, cy = ly, cz = lz;
        unsigned long long best = 0ull;
#pragma unroll
        for (int i = 0; i < 8; ++i) {
            float dx = px[i] - cx, dy = py[i] - cy, dz = pz[i] - cz;
            float d = dx * dx + dy * dy + dz * dz;
            md[i] = fminf(md[i], d);
            unsigned long long key =
                ((unsigned long long)__float_as_uint(md[i]) << 32) | (unsigned)(i * 1024 + tid);
            best = (key > best) ? key : best;
        }
#pragma unroll
        for (int off = 16; off > 0; off >>= 1) {
            unsigned long long o = shfl_xor_u64(best, off);
            best = (o > best) ? o : best;
        }
        if ((tid & 31) == 0) red[tid >> 5] = best;
        __syncthreads();
        if (tid < 32) {
            unsigned long long k = red[tid];
#pragma unroll
            for (int off = 16; off > 0; off >>= 1) {
                unsigned long long o = shfl_xor_u64(k, off);
                k = (o > k) ? o : k;
            }
            if (tid == 0) {
                int nxt = (int)(unsigned)k;
                fidx[b * G_ + s] = nxt;
                lx = p[nxt * 3 + 0];
                ly = p[nxt * 3 + 1];
                lz = p[nxt * 3 + 2];
            }
        }
        __syncthreads();
    }
}

// ---------------------------------------------------------------------------
// Kernel: KNN top-32 smallest distances per (b,g). One block (256 thr) per
// group; each thread holds 32 strided distances in registers; 32 rounds of
// block argmin with packed keys.
// ---------------------------------------------------------------------------
__global__ void knn_kernel(const float* __restrict__ xyz, const int* __restrict__ fidx,
                           int* __restrict__ knn, float* __restrict__ centers) {
    const int bg  = blockIdx.x;
    const int b   = bg >> 9;              // /G_
    const int tid = threadIdx.x;          // 256 threads
    const float* p = xyz + (size_t)b * N_ * 3;

    __shared__ float c3[3];
    __shared__ unsigned long long red[8];
    __shared__ unsigned long long winkey;

    if (tid == 0) {
        int ci = fidx[bg];
        c3[0] = p[ci * 3 + 0];
        c3[1] = p[ci * 3 + 1];
        c3[2] = p[ci * 3 + 2];
        centers[bg * 3 + 0] = c3[0];
        centers[bg * 3 + 1] = c3[1];
        centers[bg * 3 + 2] = c3[2];
    }
    __syncthreads();
    const float cx = c3[0], cy = c3[1], cz = c3[2];

    float d[32];
#pragma unroll
    for (int j = 0; j < 32; ++j) {
        int pt = j * 256 + tid;
        float dx = p[pt * 3 + 0] - cx;
        float dy = p[pt * 3 + 1] - cy;
        float dz = p[pt * 3 + 2] - cz;
        d[j] = dx * dx + dy * dy + dz * dz;
    }

    for (int s = 0; s < K_; ++s) {
        unsigned long long best = ~0ull;
#pragma unroll
        for (int j = 0; j < 32; ++j) {
            unsigned long long key =
                ((unsigned long long)__float_as_uint(d[j]) << 32) | (unsigned)(j * 256 + tid);
            best = (key < best) ? key : best;
        }
#pragma unroll
        for (int off = 16; off > 0; off >>= 1) {
            unsigned long long o = shfl_xor_u64(best, off);
            best = (o < best) ? o : best;
        }
        if ((tid & 31) == 0) red[tid >> 5] = best;
        __syncthreads();
        if (tid == 0) {
            unsigned long long wk = red[0];
#pragma unroll
            for (int w = 1; w < 8; ++w) wk = (red[w] < wk) ? red[w] : wk;
            winkey = wk;
            knn[bg * K_ + s] = (int)(unsigned)wk;
        }
        __syncthreads();
        int pt = (int)(unsigned)winkey;
#pragma unroll
        for (int j = 0; j < 32; ++j)
            if (pt == j * 256 + tid) d[j] = 3.0e38f;
        __syncthreads();
    }
}

// ---------------------------------------------------------------------------
// Kernel: build per-group 32x32 f16 feature tile (6 real channels, rest 0):
// [dx,dy,dz, r,g,b, 0...0] — zero-padded so layer-1 WMMA can use K=32.
// ---------------------------------------------------------------------------
__global__ void feats_kernel(const float* __restrict__ xyz, const float* __restrict__ color,
                             const int* __restrict__ knn, const float* __restrict__ centers,
                             _Float16* __restrict__ feats) {
    const int bg  = blockIdx.x;
    const int b   = bg >> 9;
    const int tid = threadIdx.x;          // 256 threads
    _Float16* A = feats + (size_t)bg * 32 * 32;

    for (int i = tid; i < 32 * 32; i += 256) A[i] = (_Float16)0.0f;
    __syncthreads();

    if (tid < K_) {
        int pt = knn[bg * K_ + tid];
        float cx = centers[bg * 3 + 0];
        float cy = centers[bg * 3 + 1];
        float cz = centers[bg * 3 + 2];
        const float* pp = xyz   + ((size_t)b * N_ + pt) * 3;
        const float* cc = color + ((size_t)b * N_ + pt) * 3;
        A[tid * 32 + 0] = (_Float16)(pp[0] - cx);
        A[tid * 32 + 1] = (_Float16)(pp[1] - cy);
        A[tid * 32 + 2] = (_Float16)(pp[2] - cz);
        A[tid * 32 + 3] = (_Float16)cc[0];
        A[tid * 32 + 4] = (_Float16)cc[1];
        A[tid * 32 + 5] = (_Float16)cc[2];
    }
}

// ---------------------------------------------------------------------------
// WMMA tile helpers (documented 16-bit A/B striping, wave32):
//   lane holds M (resp. N) = lane&15; K chunk base = (lane>>4)*8;
//   halves 0-7 -> K = kb..kb+7, halves 8-15 -> K = kb+16..kb+23.
// ---------------------------------------------------------------------------
#define LDA 520   // halves; row stride 1040 B (16B aligned, bank-skewed)
#define LDC 516   // floats

__device__ __forceinline__ v16h load_tile_A_lds(const _Float16* A,
                                                int m0, int k0, int lane) {
    int m  = m0 + (lane & 15);
    int kb = k0 + ((lane >> 4) << 3);
    V16 u;
    u.h[0] = *(const v8h*)(A + m * LDA + kb);
    u.h[1] = *(const v8h*)(A + m * LDA + kb + 16);
    return u.v;
}

__device__ __forceinline__ v16h load_tile_B_glb(const _Float16* __restrict__ W, int ldw,
                                                int n0, int k0, int lane) {
    int n  = n0 + (lane & 15);
    int kb = k0 + ((lane >> 4) << 3);
    V16 u;
    u.h[0] = *(const v8h*)(W + (size_t)n * ldw + kb);
    u.h[1] = *(const v8h*)(W + (size_t)n * ldw + kb + 16);
    return u.v;
}

// C/D layout: lane = N + 16*(M>=8); VGPR r holds row M = r + 8*(lane>>4).
__device__ __forceinline__ void store_C_lds(float* Cb, int m0, int n0,
                                            int lane, v8f c) {
    int n  = n0 + (lane & 15);
    int mb = m0 + ((lane >> 4) << 3);
#pragma unroll
    for (int r = 0; r < 8; ++r) Cb[(mb + r) * LDC + n] = c[r];
}

// ---------------------------------------------------------------------------
// Per-wave GEMM slab, accumulator-resident, K-outer:
//   C[32 x NT*16] slice; A tiles loaded 1x per K step (2 ds_load_b128 pairs),
//   B tiles streamed from L2. Bias folded into accumulator init.
// ---------------------------------------------------------------------------
template <int NT, int KT>
__device__ __forceinline__ void wave_gemm_t(const _Float16* A,
                                            const _Float16* __restrict__ W, int ldw,
                                            float* Cb,
                                            const float* __restrict__ bias,
                                            int n_start, int lane) {
    v8f acc[NT][2];
#pragma unroll
    for (int nt = 0; nt < NT; ++nt) {
        float bv = bias[n_start + nt * 16 + (lane & 15)];
#pragma unroll
        for (int mt = 0; mt < 2; ++mt)
#pragma unroll
            for (int r = 0; r < 8; ++r) acc[nt][mt][r] = bv;
    }
    for (int kt = 0; kt < KT; ++kt) {
        v16h a0 = load_tile_A_lds(A, 0,  kt * 32, lane);
        v16h a1 = load_tile_A_lds(A, 16, kt * 32, lane);
#pragma unroll
        for (int nt = 0; nt < NT; ++nt) {
            v16h b = load_tile_B_glb(W, ldw, n_start + nt * 16, kt * 32, lane);
            acc[nt][0] = __builtin_amdgcn_wmma_f32_16x16x32_f16(
                             false, a0, false, b, (short)0, acc[nt][0], false, false);
            acc[nt][1] = __builtin_amdgcn_wmma_f32_16x16x32_f16(
                             false, a1, false, b, (short)0, acc[nt][1], false, false);
        }
    }
#pragma unroll
    for (int nt = 0; nt < NT; ++nt) {
        store_C_lds(Cb, 0,  n_start + nt * 16, lane, acc[nt][0]);
        store_C_lds(Cb, 16, n_start + nt * 16, lane, acc[nt][1]);
    }
}

// ---------------------------------------------------------------------------
// Parallel LayerNorm + ReLU over rows: 8 threads/row, shfl_xor subgroup reduce
// ---------------------------------------------------------------------------
__device__ __forceinline__ void ln_relu_par(const float* Csh, _Float16* Ash,
                                            const float* __restrict__ gamma,
                                            const float* __restrict__ beta,
                                            int C, int tid) {
    int m   = tid >> 3;                   // 32 rows
    int sub = tid & 7;                    // 8 threads per row (same wave)
    float s = 0.f, ss = 0.f;
    for (int c = sub; c < C; c += 8) {
        float v = Csh[m * LDC + c];
        s += v; ss += v * v;
    }
#pragma unroll
    for (int off = 1; off < 8; off <<= 1) {
        s  += __shfl_xor(s, off, 32);
        ss += __shfl_xor(ss, off, 32);
    }
    float inv = 1.0f / (float)C;
    float mu  = s * inv;
    float var = ss * inv - mu * mu;
    float rs  = rsqrtf(var + 1e-5f);
    for (int c = sub; c < C; c += 8) {
        float v = (Csh[m * LDC + c] - mu) * rs * gamma[c] + beta[c];
        Ash[m * LDA + c] = (_Float16)(v > 0.f ? v : 0.f);
    }
}

// ---------------------------------------------------------------------------
// Kernel: full encoder MLP for one group per block (256 threads = 8 waves).
// ---------------------------------------------------------------------------
__global__ void encoder_kernel(const _Float16* __restrict__ feats,
                               const _Float16* __restrict__ w1p, const float* __restrict__ b1,
                               const float* __restrict__ g1,  const float* __restrict__ be1,
                               const _Float16* __restrict__ w2p, const float* __restrict__ b2,
                               const _Float16* __restrict__ w3p, const float* __restrict__ b3,
                               const float* __restrict__ g2,  const float* __restrict__ be2,
                               const _Float16* __restrict__ w4p, const float* __restrict__ b4,
                               float* __restrict__ out) {
    __shared__ _Float16 Ash[32 * LDA];    // 33,280 B: f16 activations
    __shared__ float    Csh[32 * LDC];    // 66,048 B: f32 GEMM results

    const int bg   = blockIdx.x;
    const int tid  = threadIdx.x;
    const int lane = tid & 31;
    const int wave = tid >> 5;            // 8 waves

    // ---- prefetch this wave's W3/W4 slab (64 rows x 1 KB) into cache ----
    {
        const char* p3 = (const char*)(w3p + (size_t)(wave * 64) * 512);
        const char* p4 = (const char*)(w4p + (size_t)(wave * 64) * 512);
        for (int i = lane; i < 512; i += 32) {    // 128B lines, 64 KB per wave
            __builtin_prefetch(p3 + i * 128, 0, 1);
            __builtin_prefetch(p4 + i * 128, 0, 1);
        }
    }

    // ---- stage features (32x32) into LDS ----
    const _Float16* Fg = feats + (size_t)bg * 32 * 32;
    for (int i = tid; i < 32 * 32; i += 256) {
        int m = i >> 5, k = i & 31;
        Ash[m * LDA + k] = Fg[i];
    }
    __syncthreads();

    // ---- L1: [32,32(pad 6)] x W1^T -> [32,128] ----
    wave_gemm_t<1, 1>(Ash, w1p, 32, Csh, b1, wave * 16, lane);
    __syncthreads();

    // ---- LN(128) + ReLU -> Ash f16 ----
    ln_relu_par(Csh, Ash, g1, be1, 128, tid);
    __syncthreads();

    // ---- L2: [32,128] x W2^T -> [32,256] ----
    wave_gemm_t<2, 4>(Ash, w2p, 128, Csh, b2, wave * 32, lane);
    __syncthreads();

    // ---- max over K rows -> broadcast concat: A[:,0:256]=max, A[:,256:512]=h2 ----
    {
        int c = tid;                      // 256 channels
        float mx = -3.0e38f;
        for (int m = 0; m < 32; ++m) mx = fmaxf(mx, Csh[m * LDC + c]);
        _Float16 hmx = (_Float16)mx;
        for (int m = 0; m < 32; ++m) {
            Ash[m * LDA + c]       = hmx;
            Ash[m * LDA + 256 + c] = (_Float16)Csh[m * LDC + c];
        }
    }
    __syncthreads();

    // ---- L3: [32,512] x W3^T -> [32,512] ----
    wave_gemm_t<4, 16>(Ash, w3p, 512, Csh, b3, wave * 64, lane);
    __syncthreads();

    // ---- LN(512) + ReLU -> Ash f16 ----
    ln_relu_par(Csh, Ash, g2, be2, 512, tid);
    __syncthreads();

    // ---- L4: [32,512] x W4^T -> [32,512] ----
    wave_gemm_t<4, 16>(Ash, w4p, 512, Csh, b4, wave * 64, lane);
    __syncthreads();

    // ---- max over K rows -> out[bg, 0:512] ----
    for (int c = tid; c < ENC_; c += 256) {
        float mx = -3.0e38f;
        for (int m = 0; m < 32; ++m) mx = fmaxf(mx, Csh[m * LDC + c]);
        out[(size_t)bg * ENC_ + c] = mx;
    }
}

// ---------------------------------------------------------------------------
// Launch
// ---------------------------------------------------------------------------
extern "C" void kernel_launch(void* const* d_in, const int* in_sizes, int n_in,
                              void* d_out, int out_size, void* d_ws, size_t ws_size,
                              hipStream_t stream) {
    const float* xyz   = (const float*)d_in[0];
    const float* color = (const float*)d_in[1];
    const float* w1 = (const float*)d_in[2];  const float* b1  = (const float*)d_in[3];
    const float* g1 = (const float*)d_in[4];  const float* be1 = (const float*)d_in[5];
    const float* w2 = (const float*)d_in[6];  const float* b2  = (const float*)d_in[7];
    const float* w3 = (const float*)d_in[8];  const float* b3  = (const float*)d_in[9];
    const float* g2 = (const float*)d_in[10]; const float* be2 = (const float*)d_in[11];
    const float* w4 = (const float*)d_in[12]; const float* b4  = (const float*)d_in[13];
    float* out = (float*)d_out;

    char* ws = (char*)d_ws;
    size_t off = 0;
    int*       fidx    = (int*)(ws + off);        off += (size_t)BG_ * 4;            // 16 KB
    int*       knn     = (int*)(ws + off);        off += (size_t)BG_ * K_ * 4;       // 512 KB
    float*     centers = (float*)(ws + off);      off += (size_t)BG_ * 3 * 4;        // 48 KB
    _Float16*  feats   = (_Float16*)(ws + off);   off += (size_t)BG_ * 32 * 32 * 2;  // 8 MB
    _Float16*  w1p     = (_Float16*)(ws + off);   off += (size_t)128 * 32 * 2;
    _Float16*  w2p     = (_Float16*)(ws + off);   off += (size_t)256 * 128 * 2;
    _Float16*  w3p     = (_Float16*)(ws + off);   off += (size_t)512 * 512 * 2;
    _Float16*  w4p     = (_Float16*)(ws + off);   off += (size_t)512 * 512 * 2;

    // f16 weight packing (w1 K-padded 6 -> 32)
    cvt_pad_kernel<<<(128 * 32  + 255) / 256, 256, 0, stream>>>(w1, w1p, 128, 6,   32);
    cvt_pad_kernel<<<(256 * 128 + 255) / 256, 256, 0, stream>>>(w2, w2p, 256, 128, 128);
    cvt_pad_kernel<<<(512 * 512 + 255) / 256, 256, 0, stream>>>(w3, w3p, 512, 512, 512);
    cvt_pad_kernel<<<(512 * 512 + 255) / 256, 256, 0, stream>>>(w4, w4p, 512, 512, 512);

    fps_kernel<<<B_, 1024, 0, stream>>>(xyz, fidx);
    knn_kernel<<<BG_, 256, 0, stream>>>(xyz, fidx, knn, centers);
    feats_kernel<<<BG_, 256, 0, stream>>>(xyz, color, knn, centers, feats);
    encoder_kernel<<<BG_, 256, 0, stream>>>(feats,
                                            w1p, b1, g1, be1,
                                            w2p, b2,
                                            w3p, b3, g2, be2,
                                            w4p, b4,
                                            out);
}
